// CausalAttention_18167711662612
// MI455X (gfx1250) — matrix operand order repaired and checked by reading, not compiled, not run on hardware
//
#include <hip/hip_runtime.h>

// ---------------------------------------------------------------------------
// CausalAttention for MI455X (gfx1250): bf16 WMMA pipeline, async-LDS staged.
// Workspace layout (needs ~80 MB of d_ws):
//   xb    @ 0         : 4096x2048 bf16 (x converted)              16,777,216 B
//   wqkvt @ 16777216  : 3072x2048 bf16 (Wq|Wk|Wv transposed)      12,582,912 B
//   wot   @ 29360128  : 2048x2048 bf16 (Wo transposed)             8,388,608 B
//   qkv   @ 37748736  : 4096x3072 bf16 (q|k|v, rope applied)      25,165,824 B
//   vt    @ 62914560  : (2*4)x128x2048 bf16 (V transposed)         4,194,304 B
//   attn  @ 67108864  : 4096x2048 bf16 (attention output)         16,777,216 B
// ---------------------------------------------------------------------------

#define DEVINL __device__ __forceinline__

typedef __attribute__((ext_vector_type(16))) __bf16 v16bf;
typedef __attribute__((ext_vector_type(8)))  float  v8f;
typedef __attribute__((ext_vector_type(4)))  unsigned int u32x4;

constexpr int Tn   = 2048;   // sequence length
constexpr int HID  = 2048;
constexpr int QKVC = 3072;   // 2048 q + 512 k + 512 v columns
constexpr int ROWS = 4096;   // B*T
constexpr float QK_SCALE = 0.5303300858899107f; // 6/sqrt(128)

union BFrag {
  v16bf bf;
  unsigned short us[16];
  u32x4 u4[2];
};

DEVINL unsigned short f2bf(float f) {
  union { float f; unsigned u; } v; v.f = f;
  unsigned r = v.u + 0x7fffu + ((v.u >> 16) & 1u);
  return (unsigned short)(r >> 16);
}
DEVINL float bf2f(unsigned short s) {
  union { unsigned u; float f; } v; v.u = ((unsigned)s) << 16; return v.f;
}

// A-fragment (16x32, 16-bit): lane m=lane%16 holds K in {8h..8h+7} u {16+8h..16+8h+7}
DEVINL BFrag load_a(const unsigned short* p, int lh) {
  BFrag f;
  f.u4[0] = *reinterpret_cast<const u32x4*>(p + 8 * lh);
  f.u4[1] = *reinterpret_cast<const u32x4*>(p + 16 + 8 * lh);
  return f;
}
// B-fragment (32x16, 16-bit) from row-major B^T: lane n=lane%16 holds K in {8h..8h+7} u {16+8h..16+8h+7}
DEVINL BFrag load_b(const unsigned short* p, int lh) {
  BFrag f;
  f.u4[0] = *reinterpret_cast<const u32x4*>(p + 8 * lh);
  f.u4[1] = *reinterpret_cast<const u32x4*>(p + 16 + 8 * lh);
  return f;
}

DEVINL v8f wmma_bf(const BFrag& a, const BFrag& b, v8f c) {
  return __builtin_amdgcn_wmma_f32_16x16x32_bf16(
      false, a.bf, false, b.bf, (short)0, c, false, false);
}

// CDNA5 async global->LDS copy (16 bytes per lane), tracked by ASYNCcnt.
DEVINL void async_ld_b128(unsigned lds_off, const unsigned short* gsrc) {
  asm volatile("global_load_async_to_lds_b128 %0, %1, off"
               :: "v"(lds_off), "v"(gsrc) : "memory");
}
DEVINL void wait_async0() {
  asm volatile("s_wait_asynccnt 0x0" ::: "memory");
}

// ---------------------------------------------------------------------------
// Elementwise converters
// ---------------------------------------------------------------------------
__global__ void cvt_f32_to_bf16(const float* __restrict__ in,
                                unsigned short* __restrict__ out, int n) {
  int i = blockIdx.x * blockDim.x + threadIdx.x;
  if (i < n) out[i] = f2bf(in[i]);
}

__global__ void build_wqkvt(const float* __restrict__ Wq,
                            const float* __restrict__ Wk,
                            const float* __restrict__ Wv,
                            unsigned short* __restrict__ Wt) {
  int i = blockIdx.x * blockDim.x + threadIdx.x; // over 3072*2048
  int n = i >> 11;          // output column (row of Wt)
  int k = i & 2047;         // input feature
  float v;
  if (n < 2048)      v = Wq[(size_t)k * 2048 + n];
  else if (n < 2560) v = Wk[(size_t)k * 512 + (n - 2048)];
  else               v = Wv[(size_t)k * 512 + (n - 2560)];
  Wt[i] = f2bf(v);
}

__global__ void build_wot(const float* __restrict__ Wo,
                          unsigned short* __restrict__ Wt) {
  int i = blockIdx.x * blockDim.x + threadIdx.x; // over 2048*2048
  int n = i >> 11;
  int k = i & 2047;
  Wt[i] = f2bf(Wo[(size_t)k * 2048 + n]);
}

// Vt[(b*4+kvh)*128 + d][t] = qkv[b*T+t][2560 + kvh*128 + d]
__global__ void build_vt(const unsigned short* __restrict__ qkv,
                         unsigned short* __restrict__ vt) {
  int i = blockIdx.x * blockDim.x + threadIdx.x; // over 2*4*128*2048
  int t   = i & 2047;
  int d   = (i >> 11) & 127;
  int kvh = (i >> 18) & 3;
  int b   = i >> 20;
  vt[i] = qkv[(size_t)(b * Tn + t) * QKVC + 2560 + kvh * 128 + d];
}

// In-place RoPE on q (with global scale) and k sections of qkv.
__global__ void rope_scale(unsigned short* __restrict__ qkv) {
  int idx = blockIdx.x * blockDim.x + threadIdx.x; // over 4096*20
  int row = idx / 20;
  int hh  = idx % 20;
  int t   = row & (Tn - 1);
  bool isq = hh < 16;
  unsigned short* p = isq ? qkv + (size_t)row * QKVC + hh * 128
                          : qkv + (size_t)row * QKVC + 2048 + (hh - 16) * 128;
  const float LOG1E4_O16 = 9.210340371976184f / 16.0f;
#pragma unroll
  for (int i = 0; i < 16; ++i) {
    float fr = (float)t * __expf(-(float)i * LOG1E4_O16);
    float c = __cosf(fr), s = __sinf(fr);
    float x1 = bf2f(p[i]), x2 = bf2f(p[i + 16]);
    float o1 = x1 * c - x2 * s;
    float o2 = x2 * c + x1 * s;
    if (isq) { o1 *= QK_SCALE; o2 *= QK_SCALE; }
    p[i]      = f2bf(o1);
    p[i + 16] = f2bf(o2);
  }
  if (isq) {
#pragma unroll
    for (int d = 32; d < 128; ++d) p[d] = f2bf(bf2f(p[d]) * QK_SCALE);
  }
}

// ---------------------------------------------------------------------------
// GEMM: C[MxN] = A[MxK](bf16 rowmajor) * Bt[NxK]^T (bf16).
// 256 threads = 8 waves, each wave a 16x64 tile; block = 16 rows x 512 cols.
// A tile staged to LDS via async copies (double buffered); B fragments in
// alternating register sets (no rotation movs). Barrier sits at phase TOP so
// the legalizer's pre-barrier vmem drain happens BEFORE this phase's loads
// are issued -> B loads for chunk c+1 overlap chunk c's WMMAs.
// ---------------------------------------------------------------------------
template <int OUTF32>
__global__ __launch_bounds__(256) void gemm_bf16(
    const unsigned short* __restrict__ A, const unsigned short* __restrict__ Bt,
    unsigned short* __restrict__ Cb, float* __restrict__ Cf, int N, int K) {
  __shared__ unsigned short sA[2][16 * 32];

  int tid  = threadIdx.x;
  int lane = tid & 31, wave = tid >> 5;
  int lr = lane & 15, lh = lane >> 4;
  int n0 = (blockIdx.x * 8 + wave) * 64;
  int m0 = blockIdx.y * 16;
  const int NC = K >> 5;   // K-chunks of 32; K is a multiple of 64 here

  // Staging assignment: threads 0..63 (= waves 0,1) copy 16B each.
  int srow = tid >> 2, sg = tid & 3;
  const unsigned short* aSrc = A + (size_t)(m0 + srow) * K + 8 * sg;
  unsigned lds0 = (unsigned)(size_t)&sA[0][srow * 32 + 8 * sg];
  unsigned lds1 = (unsigned)(size_t)&sA[1][srow * 32 + 8 * sg];

  if (tid < 64) async_ld_b128(lds0, aSrc);

  const unsigned short* brow = Bt + (size_t)(n0 + lr) * K;
  BFrag bA[4], bB[4];
#pragma unroll
  for (int j = 0; j < 4; ++j) bA[j] = load_b(brow + (size_t)(16 * j) * K, lh);

  v8f acc[4];
#pragma unroll
  for (int j = 0; j < 4; ++j)
#pragma unroll
    for (int r = 0; r < 8; ++r) acc[j][r] = 0.0f;

  wait_async0();

  for (int c = 0; c < NC; c += 2) {
    // ---- phase 0: compute chunk c (buf0, bA); stage/load chunk c+1 ----
    __syncthreads();
    {
      int kn = (c + 1) << 5;                 // c+1 < NC always (NC even)
      if (tid < 64) async_ld_b128(lds1, aSrc + kn);
#pragma unroll
      for (int j = 0; j < 4; ++j)
        bB[j] = load_b(brow + (size_t)(16 * j) * K + kn, lh);
      BFrag af = load_a(&sA[0][lr * 32], lh);
#pragma unroll
      for (int j = 0; j < 4; ++j) acc[j] = wmma_bf(af, bA[j], acc[j]);
      wait_async0();
    }
    // ---- phase 1: compute chunk c+1 (buf1, bB); stage/load chunk c+2 ----
    __syncthreads();
    {
      int kn = (c + 2) << 5;
      if (c + 2 < NC) {
        if (tid < 64) async_ld_b128(lds0, aSrc + kn);
#pragma unroll
        for (int j = 0; j < 4; ++j)
          bA[j] = load_b(brow + (size_t)(16 * j) * K + kn, lh);
      }
      BFrag af = load_a(&sA[1][lr * 32], lh);
#pragma unroll
      for (int j = 0; j < 4; ++j) acc[j] = wmma_bf(af, bB[j], acc[j]);
      wait_async0();
    }
  }

#pragma unroll
  for (int j = 0; j < 4; ++j)
#pragma unroll
    for (int r = 0; r < 8; ++r) {
      int row = m0 + r + 8 * lh;
      int col = n0 + 16 * j + lr;
      if (OUTF32) Cf[(size_t)row * N + col] = acc[j][r];
      else        Cb[(size_t)row * N + col] = f2bf(acc[j][r]);
    }
}

// ---------------------------------------------------------------------------
// Flash attention block step. MASK=false for key blocks strictly below the
// causal diagonal (no compares in the hot loop); MASK=true for edge blocks.
// ---------------------------------------------------------------------------
template <bool MASK>
DEVINL void attn_block(int kcol, int q0, int lr, int lh,
                       const BFrag (&qf)[4],
                       const unsigned short* kbase,
                       const unsigned short* vbase,
                       v8f (&o)[8], float& m_run, float& l_run,
                       bool doPrefetch) {
  // Upper 16-key half-tile entirely above diagonal? (wave-uniform) Skip its
  // loads and WMMAs; the mask still forces those scores to -1e30.
  bool hiActive = !MASK || (kcol + 16 <= q0 + 15);

  // Issue K fragment loads first, then V fragment loads: the S-WMMAs wait
  // only on the K loads, leaving the V loads in flight behind them.
  BFrag ka[4], kc[4];
#pragma unroll
  for (int kk = 0; kk < 4; ++kk)
    ka[kk] = load_a(kbase + (size_t)(kcol + lr) * QKVC + 32 * kk, lh);
  if (hiActive) {
#pragma unroll
    for (int kk = 0; kk < 4; ++kk)
      kc[kk] = load_a(kbase + (size_t)(kcol + 16 + lr) * QKVC + 32 * kk, lh);
  }
  BFrag vf[8];
#pragma unroll
  for (int j = 0; j < 8; ++j)
    vf[j] = load_b(vbase + (size_t)(16 * j + lr) * Tn + kcol, lh);

  // Cache prefetch of next key block (K rows + V rows).
  if (doPrefetch) {
    int lane = lr + 16 * lh;
    __builtin_prefetch(kbase + (size_t)(kcol + 32 + lane) * QKVC, 0, 3);
#pragma unroll
    for (int jj = 0; jj < 4; ++jj)
      __builtin_prefetch(vbase + (size_t)(32 * jj + lane) * Tn + kcol + 32, 0, 3);
  }

  v8f s1, s2;
#pragma unroll
  for (int r = 0; r < 8; ++r) { s1[r] = 0.0f; s2[r] = MASK ? -1e30f : 0.0f; }
#pragma unroll
  for (int kk = 0; kk < 4; ++kk) s1 = wmma_bf(ka[kk], qf[kk], s1);
  if (hiActive) {
    if (MASK) {
#pragma unroll
      for (int r = 0; r < 8; ++r) s2[r] = 0.0f;
    }
#pragma unroll
    for (int kk = 0; kk < 4; ++kk) s2 = wmma_bf(kc[kk], qf[kk], s2);
  }

  if (MASK) {
    // causal mask: lane's query = q0+lr; key rows = kcol + r + 8*lh (+16: s2)
    int qg = q0 + lr;
#pragma unroll
    for (int r = 0; r < 8; ++r) {
      if (kcol + r + 8 * lh > qg)      s1[r] = -1e30f;
      if (kcol + 16 + r + 8 * lh > qg) s2[r] = -1e30f;
    }
  }

  float mb = -1e30f;
#pragma unroll
  for (int r = 0; r < 8; ++r) mb = fmaxf(mb, fmaxf(s1[r], s2[r]));
  mb = fmaxf(mb, __shfl_xor(mb, 16, 32));
  float mnew  = fmaxf(m_run, mb);
  float alpha = __expf(m_run - mnew);
  float lb = 0.0f;
  BFrag pf;
#pragma unroll
  for (int r = 0; r < 8; ++r) {
    float p1 = __expf(s1[r] - mnew);
    float p2 = __expf(s2[r] - mnew);
    lb += p1 + p2;
    pf.us[r]     = f2bf(p1);   // A-frag elems 0..7  <-> keys kcol+8h+r
    pf.us[8 + r] = f2bf(p2);   // A-frag elems 8..15 <-> keys kcol+16+8h+r
  }
  lb += __shfl_xor(lb, 16, 32);
  l_run = l_run * alpha + lb;
  m_run = mnew;

  float ar[8];
#pragma unroll
  for (int r = 0; r < 8; ++r) ar[r] = __shfl(alpha, r + 8 * lh, 32);
#pragma unroll
  for (int j = 0; j < 8; ++j) {
#pragma unroll
    for (int r = 0; r < 8; ++r) o[j][r] *= ar[r];
    o[j] = wmma_bf(pf, vf[j], o[j]);
  }
}

// ---------------------------------------------------------------------------
// Flash attention: one wave per (b, head, 16-query tile). Computes S^T=K*Q^T
// so the exp'd scores repack into the P A-fragment in-lane (no shuffles).
// ---------------------------------------------------------------------------
__global__ __launch_bounds__(128) void flash_attn(
    const unsigned short* __restrict__ qkv,
    const unsigned short* __restrict__ vt,
    unsigned short* __restrict__ attn) {
  int lane = threadIdx.x & 31, wave = threadIdx.x >> 5;
  int lr = lane & 15, lh = lane >> 4;
  int tile = blockIdx.x * 4 + wave;      // 0..4095
  int qt  = tile & 127;
  int h   = (tile >> 7) & 15;
  int b   = tile >> 11;
  int kvh = h >> 2;
  int q0  = qt * 16;

  // Q fragments (B-operand of S^T = K * Q^T)
  const unsigned short* qrow = qkv + (size_t)(b * Tn + q0 + lr) * QKVC + h * 128;
  BFrag qf[4];
#pragma unroll
  for (int kk = 0; kk < 4; ++kk) qf[kk] = load_b(qrow + 32 * kk, lh);

  const unsigned short* kbase = qkv + (size_t)(b * Tn) * QKVC + 2048 + kvh * 128;
  const unsigned short* vbase = vt + (size_t)(b * 4 + kvh) * 128 * Tn;

  v8f o[8];
#pragma unroll
  for (int j = 0; j < 8; ++j)
#pragma unroll
    for (int r = 0; r < 8; ++r) o[j][r] = 0.0f;

  float m_run = -1e30f, l_run = 0.0f;
  int nkb   = (q0 + 47) >> 5; // ceil((q0+16)/32) blocks of 32 keys
  int nfull = q0 >> 5;        // blocks strictly below the diagonal

  int kb = 0;
  for (; kb < nfull; ++kb)
    attn_block<false>(kb * 32, q0, lr, lh, qf, kbase, vbase,
                      o, m_run, l_run, kb + 1 < nkb);
  for (; kb < nkb; ++kb)
    attn_block<true>(kb * 32, q0, lr, lh, qf, kbase, vbase,
                     o, m_run, l_run, kb + 1 < nkb);

  float inv = 1.0f / l_run;
  float ir[8];
#pragma unroll
  for (int r = 0; r < 8; ++r) ir[r] = __shfl(inv, r + 8 * lh, 32);
  unsigned short* orow = attn + (size_t)(b * Tn + q0) * HID + h * 128;
#pragma unroll
  for (int j = 0; j < 8; ++j)
#pragma unroll
    for (int r = 0; r < 8; ++r)
      orow[(size_t)(r + 8 * lh) * HID + 16 * j + lr] = f2bf(o[j][r] * ir[r]);
}

// ---------------------------------------------------------------------------
extern "C" void kernel_launch(void* const* d_in, const int* in_sizes, int n_in,
                              void* d_out, int out_size, void* d_ws, size_t ws_size,
                              hipStream_t stream) {
  const float* x  = (const float*)d_in[0];
  const float* Wq = (const float*)d_in[1];
  const float* Wk = (const float*)d_in[2];
  const float* Wv = (const float*)d_in[3];
  const float* Wo = (const float*)d_in[4];
  float* out = (float*)d_out;

  char* ws = (char*)d_ws;
  unsigned short* xb    = (unsigned short*)(ws + 0);
  unsigned short* wqkvt = (unsigned short*)(ws + 16777216);
  unsigned short* wot   = (unsigned short*)(ws + 29360128);
  unsigned short* qkv   = (unsigned short*)(ws + 37748736);
  unsigned short* vtb   = (unsigned short*)(ws + 62914560);
  unsigned short* attn  = (unsigned short*)(ws + 67108864);

  // 1) convert inputs to bf16 (weights transposed, qkv weights fused)
  cvt_f32_to_bf16<<<(ROWS * HID) / 256, 256, 0, stream>>>(x, xb, ROWS * HID);
  build_wqkvt<<<(QKVC * HID) / 256, 256, 0, stream>>>(Wq, Wk, Wv, wqkvt);
  build_wot<<<(HID * HID) / 256, 256, 0, stream>>>(Wo, wot);

  // 2) fused QKV projection: (4096x2048)*(2048x3072) -> bf16 qkv
  gemm_bf16<0><<<dim3(QKVC / 512, ROWS / 16), 256, 0, stream>>>(
      xb, wqkvt, qkv, nullptr, QKVC, HID);

  // 3) rope (+q scale) in place; 4) V transpose
  rope_scale<<<(ROWS * 20) / 128, 128, 0, stream>>>(qkv);
  build_vt<<<(2 * 4 * 128 * Tn) / 256, 256, 0, stream>>>(qkv, vtb);

  // 5) flash attention (one wave per 16-query tile)
  flash_attn<<<(2 * 16 * (Tn / 16)) / 4, 128, 0, stream>>>(qkv, vtb, attn);

  // 6) output projection -> f32 out
  gemm_bf16<1><<<dim3(HID / 512, ROWS / 16), 256, 0, stream>>>(
      attn, wot, nullptr, out, ROWS, HID);
}
